// simple_attention_39633958207667
// MI455X (gfx1250) — compile-verified, matrix operand order checked by reading, not run on hardware
//
#include <hip/hip_runtime.h>
#include <hip/hip_bf16.h>

// ---------------------------------------------------------------------------
// Problem constants
// ---------------------------------------------------------------------------
#define N_ROWS 65536
#define C_IN   256
#define HD     512          // H*D
#define NHEAD  8
#define DHEAD  64
static const size_t NS = (size_t)N_ROWS * HD;   // elements of qs/ks/vs
static const size_t XS = (size_t)N_ROWS * C_IN; // elements of Xq/Xs
static const int    WS1 = HD * C_IN;            // elements of one weight matrix

typedef __attribute__((ext_vector_type(16))) __bf16 v16bf;
typedef __attribute__((ext_vector_type(8)))  __bf16 v8bf;
typedef __attribute__((ext_vector_type(4)))  __bf16 v4bf;
typedef __attribute__((ext_vector_type(8)))  float  v8f;
typedef __attribute__((ext_vector_type(4)))  unsigned int v4u;
typedef __attribute__((ext_vector_type(8)))  int    v8i;
typedef __attribute__((ext_vector_type(4)))  int    v4i;

#if __has_builtin(__builtin_amdgcn_tensor_load_to_lds) && \
    __has_builtin(__builtin_amdgcn_s_wait_tensorcnt)
#define HAVE_TDM 1
#else
#define HAVE_TDM 0
#endif

__device__ __forceinline__ v8f wmma_bf16(v16bf a, v16bf b, v8f c) {
    return __builtin_amdgcn_wmma_f32_16x16x32_bf16(
        false, a, false, b, (short)0, c, false, false);
}

// 16x32 operand fragment from bf16 memory (LDS/global), ISA 16-bit layout.
__device__ __forceinline__ v16bf frag_bf16(const __bf16* __restrict__ p) {
    v8bf lo = *(const v8bf*)(p);
    v8bf hi = *(const v8bf*)(p + 16);
    v16bf f;
#pragma unroll
    for (int i = 0; i < 8; ++i) { f[i] = lo[i]; f[8 + i] = hi[i]; }
    return f;
}

// Workgroup-relative LDS byte address (low 32 bits of the generic address).
__device__ __forceinline__ unsigned lds_off(const void* p) {
    return (unsigned)(unsigned long long)p;
}

#if HAVE_TDM
// Issue one TDM load of a 2-D bf16 tile (tile_d0 x tile_d1, row stride
// stride0 elements) from global to LDS. Descriptor per ISA 8.3/8.4:
// group0: count=1, lds_addr, 57-bit global addr, type=2
// group1: data_size=1(2B), tensor dims, tile dims, dim0 stride
__device__ __forceinline__ void tdm_load_2d(unsigned lds_byte,
                                            const void* gptr,
                                            unsigned tensor_d0, unsigned tensor_d1,
                                            unsigned stride0,
                                            unsigned tile_d0, unsigned tile_d1)
{
    unsigned long long ga = (unsigned long long)gptr;
    v4u g0 = { 1u,                                   // count=1
               lds_byte,
               (unsigned)(ga & 0xFFFFFFFFu),
               (unsigned)((ga >> 32) & 0x01FFFFFFu) | (2u << 30) };  // type=2
    v8i g1 = { (int)(1u << 16),                      // data_size = 2 bytes
               (int)((tensor_d0 & 0xFFFFu) << 16),
               (int)((tensor_d0 >> 16) | ((tensor_d1 & 0xFFFFu) << 16)),
               (int)((tensor_d1 >> 16) | (tile_d0 << 16)),
               (int)tile_d1,                         // tile_dim2 = 0
               (int)stride0,                         // dim0 stride [31:0]
               0,                                    // stride0[47:32]=0, stride1 lo=0
               0 };
    v4i gz = { 0, 0, 0, 0 };
#if defined(__clang_major__) && (__clang_major__ >= 23)
    v8i gz8 = { 0, 0, 0, 0, 0, 0, 0, 0 };
    __builtin_amdgcn_tensor_load_to_lds(g0, g1, gz, gz, gz8, 0);
#else
    __builtin_amdgcn_tensor_load_to_lds(g0, g1, gz, gz, 0);
#endif
}
__device__ __forceinline__ void tdm_wait0() {
    __builtin_amdgcn_s_wait_tensorcnt(0);
}
#endif

// ---------------------------------------------------------------------------
// 0a) zero scratch accumulators
// ---------------------------------------------------------------------------
__global__ void zero_kernel(float* __restrict__ p, int n) {
    int i = blockIdx.x * 256 + threadIdx.x;
    if (i < n) p[i] = 0.0f;
}

// 0b) one-time f32 -> bf16 conversion (X matrices + weights); n % 4 == 0
__global__ void cvt_kernel(const float* __restrict__ src, __bf16* __restrict__ dst,
                           unsigned n) {
    unsigned i = (blockIdx.x * 256u + threadIdx.x) * 4u;
    if (i < n) {
        float4 f = *(const float4*)(src + i);
        v4bf o;
        o[0] = (__bf16)f.x; o[1] = (__bf16)f.y;
        o[2] = (__bf16)f.z; o[3] = (__bf16)f.w;
        *(v4bf*)(dst + i) = o;
    }
}

// ---------------------------------------------------------------------------
// 1) projections: Out[N,512](bf16) = X[N,256] @ W[512,256]^T + bias
//    grid = (row groups of 512 rows = 128, col quarter = 4, matrix = 3)
//    W slice (128x256 bf16, 64KB) loaded to LDS via one TDM op; A tiles
//    (16x256 bf16, contiguous) double-buffered via TDM issued by wave 0,
//    synchronized with s_wait_tensorcnt + workgroup barrier.
// ---------------------------------------------------------------------------
__global__ __launch_bounds__(256) void proj_kernel(
    const __bf16* __restrict__ Xqb, const __bf16* __restrict__ Xsb,
    const __bf16* __restrict__ Wb,                      // 3 matrices, contiguous
    const float* __restrict__ Bq, const float* __restrict__ Bk,
    const float* __restrict__ Bv,
    __bf16* __restrict__ qs, __bf16* __restrict__ ks, __bf16* __restrict__ vs)
{
    __shared__ __align__(16) __bf16 lw[128 * C_IN];      // 64 KB  [localcol][k]
    __shared__ __align__(16) __bf16 la[2 * 16 * C_IN];   // 16 KB  double buffer
    __shared__ float lbias[128];

    const int mat = blockIdx.z;
    const __bf16* __restrict__ X = (mat == 0) ? Xqb : Xsb;
    const __bf16* __restrict__ W = Wb + (size_t)mat * WS1;
    const float* __restrict__ Bi = (mat == 0) ? Bq : (mat == 1 ? Bk : Bv);
    __bf16* __restrict__ Out     = (mat == 0) ? qs : (mat == 1 ? ks : vs);

    const int col0 = blockIdx.y * 128;
    const int wave = threadIdx.x >> 5;
    const int lane = threadIdx.x & 31;
    const int l15  = lane & 15;
    const int kh8  = (lane >> 4) * 8;
    const int row00 = blockIdx.x * 32 * 16;

#if HAVE_TDM
    if (threadIdx.x < 32) {
        // whole W slice: rows are full -> contiguous; 2-D tile 256x128
        tdm_load_2d(lds_off(lw), W + (size_t)col0 * C_IN,
                    C_IN, HD, C_IN, C_IN, 128);
        // first A tile (16 rows x 256): contiguous, 2-D tile 256x16
        tdm_load_2d(lds_off(la), X + (size_t)row00 * C_IN,
                    C_IN, N_ROWS, C_IN, C_IN, 16);
    }
#else
    for (int j = threadIdx.x * 8; j < 128 * C_IN; j += 256 * 8)
        *(v8bf*)(lw + j) = *(const v8bf*)(W + (size_t)col0 * C_IN + j);
#endif
    if (threadIdx.x < 128) lbias[threadIdx.x] = Bi[col0 + threadIdx.x];

    const int lcol = wave * 16 + l15;                // local column (B operand)
    const int gcol = col0 + lcol;

    for (int rt = 0; rt < 32; ++rt) {
        const int row0 = row00 + rt * 16;
        const __bf16* labuf = la + (rt & 1) * (16 * C_IN);

#if HAVE_TDM
        if (threadIdx.x < 32) tdm_wait0();           // tile rt (and W) arrived
        __syncthreads();
        if (rt + 1 < 32 && threadIdx.x < 32)         // async-load tile rt+1
            tdm_load_2d(lds_off(la + ((rt + 1) & 1) * (16 * C_IN)),
                        X + (size_t)(row0 + 16) * C_IN,
                        C_IN, N_ROWS, C_IN, C_IN, 16);
#else
        __syncthreads();
        for (int j = threadIdx.x * 8; j < 16 * C_IN; j += 256 * 8)
            *(v8bf*)((__bf16*)labuf + j) =
                *(const v8bf*)(X + (size_t)row0 * C_IN + j);
        __syncthreads();
#endif

        v8f acc = {};
#pragma unroll
        for (int kk = 0; kk < C_IN; kk += 32) {
            v16bf a = frag_bf16(labuf + l15 * C_IN + kk + kh8);
            v16bf b = frag_bf16(lw + lcol * C_IN + kk + kh8);
            acc = wmma_bf16(a, b, acc);
        }

        const float bb  = lbias[lcol];
        const int mbase = row0 + (lane >> 4) * 8;
#pragma unroll
        for (int r = 0; r < 8; ++r)
            Out[(size_t)(mbase + r) * HD + gcol] = (__bf16)(acc[r] + bb);

#if !HAVE_TDM
        __syncthreads();
#endif
    }
}

// ---------------------------------------------------------------------------
// 2) reductions over bf16 qs/ks/vs
// ---------------------------------------------------------------------------
__global__ __launch_bounds__(256) void reduce_kernel(
    const __bf16* __restrict__ qs, const __bf16* __restrict__ ks,
    const __bf16* __restrict__ vs,
    float* __restrict__ ssq, float* __restrict__ ks_sum, float* __restrict__ vs_sum)
{
    __shared__ float lsk[HD];
    __shared__ float lsv[HD];
    __shared__ float red[256];

    for (int i = threadIdx.x; i < HD; i += 256) { lsk[i] = 0.0f; lsv[i] = 0.0f; }
    __syncthreads();

    float sq = 0.0f, sk = 0.0f;
    const size_t stride = (size_t)gridDim.x * 256 * 4;
    for (size_t i = ((size_t)blockIdx.x * 256 + threadIdx.x) * 4; i < NS; i += stride) {
        v4bf q4 = *(const v4bf*)(qs + i);
        v4bf k4 = *(const v4bf*)(ks + i);
        v4bf v4 = *(const v4bf*)(vs + i);
        const int c0 = (int)(i & (HD - 1));
#pragma unroll
        for (int u = 0; u < 4; ++u) {
            float q = (float)q4[u]; sq += q * q;
            float k = (float)k4[u]; sk += k * k;
            atomicAdd(&lsk[c0 + u], k);
            atomicAdd(&lsv[c0 + u], (float)v4[u]);
        }
    }

    red[threadIdx.x] = sq; __syncthreads();
    for (int s = 128; s > 0; s >>= 1) {
        if (threadIdx.x < s) red[threadIdx.x] += red[threadIdx.x + s];
        __syncthreads();
    }
    if (threadIdx.x == 0) atomicAdd(&ssq[0], red[0]);
    __syncthreads();

    red[threadIdx.x] = sk; __syncthreads();
    for (int s = 128; s > 0; s >>= 1) {
        if (threadIdx.x < s) red[threadIdx.x] += red[threadIdx.x + s];
        __syncthreads();
    }
    if (threadIdx.x == 0) atomicAdd(&ssq[1], red[0]);
    __syncthreads();

    for (int i = threadIdx.x; i < HD; i += 256) {
        atomicAdd(&ks_sum[i], lsk[i]);
        atomicAdd(&vs_sum[i], lsv[i]);
    }
}

// ---------------------------------------------------------------------------
// 3) alpha = 1 / (||qs|| * ||ks||)
// ---------------------------------------------------------------------------
__global__ void finalize_kernel(const float* __restrict__ ssq, float* __restrict__ alpha) {
    if (threadIdx.x == 0 && blockIdx.x == 0)
        alpha[0] = rsqrtf(ssq[0]) * rsqrtf(ssq[1]);
}

// ---------------------------------------------------------------------------
// 4) kvs[h][m][d] += sum_l ks[l][h*64+m] * vs[l][h*64+d]
//    32-row stages written pre-transposed to LDS (padded stride 40 halves).
// ---------------------------------------------------------------------------
#define TSTR 40
__global__ __launch_bounds__(256) void kvs_kernel(
    const __bf16* __restrict__ ks, const __bf16* __restrict__ vs,
    float* __restrict__ kvs)
{
    __shared__ __align__(16) __bf16 lkT[DHEAD * TSTR];
    __shared__ __align__(16) __bf16 lvT[DHEAD * TSTR];

    const int h      = blockIdx.y;
    const int chunk0 = blockIdx.x * 2048;
    const int wave   = threadIdx.x >> 5;
    const int lane   = threadIdx.x & 31;
    const int l15    = lane & 15;
    const int kh8    = (lane >> 4) * 8;

    const int t0 = wave, t1 = wave + 8;
    const int m0 = (t0 >> 2) * 16 + l15, d0 = (t0 & 3) * 16 + l15;
    const int m1 = (t1 >> 2) * 16 + l15, d1 = (t1 & 3) * 16 + l15;

    const int sr  = threadIdx.x >> 3;
    const int sc0 = (threadIdx.x & 7) * 8;

    v8f acc0 = {}, acc1 = {};

    for (int l0 = chunk0; l0 < chunk0 + 2048; l0 += 32) {
        __syncthreads();
        {
            const size_t src = (size_t)(l0 + sr) * HD + h * DHEAD + sc0;
            v8bf kq = *(const v8bf*)(ks + src);
            v8bf vq = *(const v8bf*)(vs + src);
            if (l0 + 32 < chunk0 + 2048)
                __builtin_prefetch(ks + src + 32 * HD, 0, 0);
#pragma unroll
            for (int i = 0; i < 8; ++i) {
                lkT[(sc0 + i) * TSTR + sr] = kq[i];
                lvT[(sc0 + i) * TSTR + sr] = vq[i];
            }
        }
        __syncthreads();

        v16bf a0 = frag_bf16(lkT + m0 * TSTR + kh8);
        v16bf b0 = frag_bf16(lvT + d0 * TSTR + kh8);
        v16bf a1 = frag_bf16(lkT + m1 * TSTR + kh8);
        v16bf b1 = frag_bf16(lvT + d1 * TSTR + kh8);
        acc0 = wmma_bf16(a0, b0, acc0);
        acc1 = wmma_bf16(a1, b1, acc1);
    }

    const int rbase = (lane >> 4) * 8;
#pragma unroll
    for (int r = 0; r < 8; ++r) {
        atomicAdd(&kvs[h * 4096 + ((t0 >> 2) * 16 + rbase + r) * DHEAD + d0], acc0[r]);
        atomicAdd(&kvs[h * 4096 + ((t1 >> 2) * 16 + rbase + r) * DHEAD + d1], acc1[r]);
    }
}

// ---------------------------------------------------------------------------
// 5) out[n][h][d] = (alpha*(qs_row . kvs[h][:,d]) + vs_sum) /
//                   (alpha*(qs_row . ks_sum[h]) + N)
//    qs tile staged via TDM; kvs staged transposed as bf16.
// ---------------------------------------------------------------------------
__global__ __launch_bounds__(256) void out_kernel(
    const __bf16* __restrict__ qs, const float* __restrict__ kvs,
    const float* __restrict__ ks_sum, const float* __restrict__ vs_sum,
    const float* __restrict__ alpha_p, float* __restrict__ out)
{
    __shared__ __align__(16) __bf16 lq[16 * HD];                // 16 KB
    __shared__ __align__(16) __bf16 kvT[NHEAD * DHEAD * DHEAD]; // 64 KB [h][d][m]
    __shared__ float lks[HD];
    __shared__ float lvs[HD];

    const int n0 = blockIdx.x * 16;
#if HAVE_TDM
    if (threadIdx.x < 32)   // 16x512 bf16 tile, contiguous
        tdm_load_2d(lds_off(lq), qs + (size_t)n0 * HD, HD, N_ROWS, HD, HD, 16);
#else
    for (int j = threadIdx.x * 8; j < 16 * HD; j += 256 * 8)
        *(v8bf*)(lq + j) = *(const v8bf*)(qs + (size_t)n0 * HD + j);
#endif
    for (int j = threadIdx.x; j < NHEAD * DHEAD * DHEAD; j += 256) {
        const int hh = j >> 12, m = (j >> 6) & 63, d = j & 63;
        kvT[(hh << 12) + (d << 6) + m] = (__bf16)kvs[j];
    }
    for (int j = threadIdx.x; j < HD; j += 256) {
        lks[j] = ks_sum[j];
        lvs[j] = vs_sum[j];
    }
#if HAVE_TDM
    if (threadIdx.x < 32) tdm_wait0();
#endif
    __syncthreads();

    const float alpha = alpha_p[0];
    const int h     = threadIdx.x >> 5;     // wave == head
    const int lane  = threadIdx.x & 31;
    const int l15   = lane & 15;
    const int kh8   = (lane >> 4) * 8;
    const int rbase = (lane >> 4) * 8;

    float rdenom[8];
#pragma unroll
    for (int r = 0; r < 8; ++r) {
        const int row = rbase + r;
        float dot = 0.0f;
#pragma unroll 8
        for (int m = 0; m < DHEAD; ++m)
            dot += (float)lq[row * HD + h * DHEAD + m] * lks[h * DHEAD + m];
        rdenom[r] = 1.0f / (alpha * dot + (float)N_ROWS);
    }

#pragma unroll
    for (int dt = 0; dt < 4; ++dt) {
        const int d = dt * 16 + l15;
        v8f acc = {};
#pragma unroll
        for (int kk = 0; kk < DHEAD; kk += 32) {
            v16bf a = frag_bf16(lq + l15 * HD + h * DHEAD + kk + kh8);
            v16bf b = frag_bf16(kvT + (h << 12) + (d << 6) + kk + kh8);
            acc = wmma_bf16(a, b, acc);
        }
#pragma unroll
        for (int r = 0; r < 8; ++r) {
            const int n = n0 + rbase + r;
            out[(size_t)n * HD + h * DHEAD + d] =
                (alpha * acc[r] + lvs[h * DHEAD + d]) * rdenom[r];
        }
    }
}

// ---------------------------------------------------------------------------
// host launcher
// ---------------------------------------------------------------------------
extern "C" void kernel_launch(void* const* d_in, const int* in_sizes, int n_in,
                              void* d_out, int out_size, void* d_ws, size_t ws_size,
                              hipStream_t stream) {
    const float* Xq = (const float*)d_in[0];
    const float* Xs = (const float*)d_in[1];
    const float* Wq = (const float*)d_in[2];
    const float* Bq = (const float*)d_in[3];
    const float* Wk = (const float*)d_in[4];
    const float* Bk = (const float*)d_in[5];
    const float* Wv = (const float*)d_in[6];
    const float* Bv = (const float*)d_in[7];
    float* out = (float*)d_out;                 // [N, 512] f32

    __bf16* qs  = (__bf16*)d_ws;                // NS halves
    __bf16* ks  = qs + NS;
    __bf16* vs  = ks + NS;
    __bf16* Xqb = vs + NS;                      // XS halves
    __bf16* Xsb = Xqb + XS;
    __bf16* Wb  = Xsb + XS;                     // 3*WS1 halves
    float* kvsb  = (float*)(Wb + 3 * (size_t)WS1);
    float* kssum = kvsb + NHEAD * DHEAD * DHEAD;
    float* vssum = kssum + HD;
    float* ssq   = vssum + HD;
    float* alpha = ssq + 2;

    // 0) zero accumulators + one-time bf16 conversions
    const int nzero = NHEAD * DHEAD * DHEAD + HD + HD + 2 + 1;
    zero_kernel<<<(nzero + 255) / 256, 256, 0, stream>>>(kvsb, nzero);
    cvt_kernel<<<(unsigned)(XS / 4 / 256), 256, 0, stream>>>(Xq, Xqb, (unsigned)XS);
    cvt_kernel<<<(unsigned)(XS / 4 / 256), 256, 0, stream>>>(Xs, Xsb, (unsigned)XS);
    cvt_kernel<<<WS1 / 4 / 256, 256, 0, stream>>>(Wq, Wb,           WS1);
    cvt_kernel<<<WS1 / 4 / 256, 256, 0, stream>>>(Wk, Wb + WS1,     WS1);
    cvt_kernel<<<WS1 / 4 / 256, 256, 0, stream>>>(Wv, Wb + 2 * WS1, WS1);

    // 1) projections
    proj_kernel<<<dim3(N_ROWS / (32 * 16), 4, 3), 256, 0, stream>>>(
        Xqb, Xsb, Wb, Bq, Bk, Bv, qs, ks, vs);

    // 2) reductions
    reduce_kernel<<<512, 256, 0, stream>>>(qs, ks, vs, ssq, kssum, vssum);

    // 3) alpha
    finalize_kernel<<<1, 64, 0, stream>>>(ssq, alpha);

    // 4) kvs accumulation
    kvs_kernel<<<dim3(32, NHEAD), 256, 0, stream>>>(ks, vs, kvsb);

    // 5) output
    out_kernel<<<N_ROWS / 16, 256, 0, stream>>>(qs, kvsb, kssum, vssum, alpha, out);
}
